// flash_attention_19387482375077
// MI455X (gfx1250) — compile-verified, matrix operand order checked by reading, not run on hardware
//
#include <hip/hip_runtime.h>

typedef __attribute__((ext_vector_type(16))) __bf16 v16bf;
typedef __attribute__((ext_vector_type(8)))  __bf16 v8bf;
typedef __attribute__((ext_vector_type(8)))  float  v8f;

#define DIMD 128
#define GRP  2048
#define ROWS_PER_BLOCK 64
#define KV_TILE 32

__device__ __forceinline__ float rowmax16(float v) {
  v = fmaxf(v, __shfl_xor(v, 1, 32));
  v = fmaxf(v, __shfl_xor(v, 2, 32));
  v = fmaxf(v, __shfl_xor(v, 4, 32));
  v = fmaxf(v, __shfl_xor(v, 8, 32));
  return v;
}
__device__ __forceinline__ float rowsum16(float v) {
  v += __shfl_xor(v, 1, 32);
  v += __shfl_xor(v, 2, 32);
  v += __shfl_xor(v, 4, 32);
  v += __shfl_xor(v, 8, 32);
  return v;
}

__device__ __forceinline__ v8bf pack8(float4 a, float4 b) {
  v8bf r;
  r[0] = (__bf16)a.x; r[1] = (__bf16)a.y; r[2] = (__bf16)a.z; r[3] = (__bf16)a.w;
  r[4] = (__bf16)b.x; r[5] = (__bf16)b.y; r[6] = (__bf16)b.z; r[7] = (__bf16)b.w;
  return r;
}

__device__ __forceinline__ v16bf combine16(v8bf lo, v8bf hi) {
  v16bf r;
#pragma unroll
  for (int i = 0; i < 8; ++i) { r[i] = lo[i]; r[i + 8] = hi[i]; }
  return r;
}

// 16-element bf16 fragment as two aligned 16-byte LDS reads.
__device__ __forceinline__ v16bf frag16(const __bf16* p0, const __bf16* p1) {
  v8bf lo = *reinterpret_cast<const v8bf*>(p0);
  v8bf hi = *reinterpret_cast<const v8bf*>(p1);
  return combine16(lo, hi);
}

__device__ __forceinline__ v8f vzero8() {
  v8f r;
#pragma unroll
  for (int i = 0; i < 8; ++i) r[i] = 0.0f;
  return r;
}

__global__ __launch_bounds__(128) void fa_causal_grouped_bf16(
    const float* __restrict__ q, const float* __restrict__ k,
    const float* __restrict__ v, float* __restrict__ out) {
  // K tile row-major [key][d]. V tile transposed [d][slot] and P scratch
  // [row][slot] use a shared K-dim relabeling: slot 2j <- key j, slot 2j+1 <-
  // key j+16 (legal: both WMMA operands of O=PV use the same permutation).
  // This makes each lane's two P values adjacent -> single b32 store.
  __shared__ __align__(16) __bf16 Klds[KV_TILE * DIMD];   // 8 KB
  __shared__ __align__(16) __bf16 VT[DIMD * KV_TILE];     // 8 KB
  __shared__ __align__(16) __bf16 Plds[4][16][KV_TILE];   // 4 KB

  const int tid  = threadIdx.x;
  const int wav  = tid >> 5;        // wave32
  const int lane = tid & 31;
  const int half = lane >> 4;       // 0: lanes 0-15, 1: lanes 16-31
  const int lq   = lane & 15;
  const int r0   = half * 8;        // per-lane K-run base inside a 32-chunk

  const int tilesPerGroup = GRP / ROWS_PER_BLOCK;           // 32
  const int g     = blockIdx.x / tilesPerGroup;
  const int rt    = blockIdx.x % tilesPerGroup;
  const int row0  = rt * ROWS_PER_BLOCK;                    // block's first Q row
  const int qbase = row0 + 16 * wav;                        // wave's first Q row

  const size_t gbase = (size_t)g * GRP * DIMD;

  // ---- Q as 4 bf16 A-fragments (16 rows x K=32 each), rows = qbase..+15 ----
  v16bf qa[4];
  {
    const float* qrow = q + gbase + (size_t)(qbase + lq) * DIMD;
#pragma unroll
    for (int c = 0; c < 4; ++c) {
      const float* p0 = qrow + 32 * c + r0;
      const float* p1 = qrow + 32 * c + r0 + 16;
      float4 a0 = reinterpret_cast<const float4*>(p0)[0];
      float4 a1 = reinterpret_cast<const float4*>(p0)[1];
      float4 b0 = reinterpret_cast<const float4*>(p1)[0];
      float4 b1 = reinterpret_cast<const float4*>(p1)[1];
      qa[c] = combine16(pack8(a0, a1), pack8(b0, b1));
    }
  }

  // ---- accumulators / running softmax state (per C-tile row r) ----
  v8f oacc[8];
#pragma unroll
  for (int d = 0; d < 8; ++d) oacc[d] = vzero8();
  float rm[8], rs[8];
#pragma unroll
  for (int r = 0; r < 8; ++r) { rm[r] = -1e30f; rs[r] = 0.0f; }

  const int nt = (row0 + ROWS_PER_BLOCK) / KV_TILE;  // causal tile bound

  for (int t = 0; t < nt; ++t) {
    const int kv0 = t * KV_TILE;

    __syncthreads();  // previous tile fully consumed before overwrite

    // ---- cooperative K/V tile load: thread = (key, d-quarter) ----
    {
      const int key  = tid & 31;
      const int d0   = (tid >> 5) * 32;
      const int vpos = 2 * (key & 15) + (key >> 4);  // permuted key slot
      const float* krow = k + gbase + (size_t)(kv0 + key) * DIMD + d0;
      const float* vrow = v + gbase + (size_t)(kv0 + key) * DIMD + d0;
#pragma unroll
      for (int j = 0; j < 4; ++j) {
        float4 f0 = reinterpret_cast<const float4*>(krow)[2 * j];
        float4 f1 = reinterpret_cast<const float4*>(krow)[2 * j + 1];
        *reinterpret_cast<v8bf*>(&Klds[key * DIMD + d0 + 8 * j]) = pack8(f0, f1);
      }
#pragma unroll
      for (int j = 0; j < 4; ++j) {
        float4 f0 = reinterpret_cast<const float4*>(vrow)[2 * j];
        float4 f1 = reinterpret_cast<const float4*>(vrow)[2 * j + 1];
        float va[8] = {f0.x, f0.y, f0.z, f0.w, f1.x, f1.y, f1.z, f1.w};
#pragma unroll
        for (int i = 0; i < 8; ++i)
          VT[(d0 + 8 * j + i) * KV_TILE + vpos] = (__bf16)va[i];
      }
      if (t + 1 < nt) {  // warm L2 for next tile (global_prefetch_b8)
        __builtin_prefetch(krow + KV_TILE * DIMD, 0, 1);
        __builtin_prefetch(vrow + KV_TILE * DIMD, 0, 1);
      }
    }
    __syncthreads();

    // Waves entirely above this KV tile skip compute (barriers stay outside).
    if (kv0 <= qbase + 15) {
      // ---- S = Q K^T : two 16x16 C-tiles (key halves), 1-deep pipelined ----
      auto kfrag = [&](int i) {
        const int h = i >> 2, c = i & 3;
        const __bf16* p = &Klds[(16 * h + lq) * DIMD + 32 * c + r0];
        return frag16(p, p + 16);
      };
      v8f st[2];
      {
        v8f sA = vzero8(), sB = vzero8();
        v16bf kb = kfrag(0);
#pragma unroll
        for (int i = 0; i < 8; ++i) {
          v16bf nxt = kb;
          if (i < 7) nxt = kfrag(i + 1);           // issue next loads early
          if (i < 4)
            sA = __builtin_amdgcn_wmma_f32_16x16x32_bf16(false, qa[i & 3], false, kb,
                                                         (short)0, sA, false, false);
          else
            sB = __builtin_amdgcn_wmma_f32_16x16x32_bf16(false, qa[i & 3], false, kb,
                                                         (short)0, sB, false, false);
          kb = nxt;
        }
        st[0] = sA; st[1] = sB;
      }

      // ---- scale by 1/sqrt(D) in f32, then causal mask ----
      const float scl = 0.08838834764831845f;  // 1/sqrt(128)
#pragma unroll
      for (int r = 0; r < 8; ++r) { st[0][r] *= scl; st[1][r] *= scl; }

      if (kv0 + KV_TILE - 1 > qbase) {  // tile crosses the diagonal
        const int rowb = qbase + r0;
#pragma unroll
        for (int r = 0; r < 8; ++r) {
          if (kv0 + lq      > rowb + r) st[0][r] = -1e30f;
          if (kv0 + 16 + lq > rowb + r) st[1][r] = -1e30f;
        }
      }

      // ---- online softmax (per-row stats via 16-lane butterflies) ----
#pragma unroll
      for (int r = 0; r < 8; ++r) {
        float tm = rowmax16(fmaxf(st[0][r], st[1][r]));
        float mn = fmaxf(rm[r], tm);
        float sc = __expf(rm[r] - mn);
        rm[r] = mn;
        float p0 = __expf(st[0][r] - mn);
        float p1 = __expf(st[1][r] - mn);
        rs[r] = rs[r] * sc + rowsum16(p0 + p1);
#pragma unroll
        for (int d = 0; d < 8; ++d) oacc[d][r] *= sc;
        // adjacent permuted slots -> one b32 store of a packed bf16 pair
        unsigned short b0 = __builtin_bit_cast(unsigned short, (__bf16)p0);
        unsigned short b1 = __builtin_bit_cast(unsigned short, (__bf16)p1);
        unsigned int pk = (unsigned int)b0 | ((unsigned int)b1 << 16);
        *reinterpret_cast<unsigned int*>(&Plds[wav][r0 + r][2 * lq]) = pk;
      }

      // Same-wave LDS store->load ordering (split dep counters on gfx1250)
      asm volatile("s_wait_dscnt 0" ::: "memory");

      // ---- O += P V : P as A-fragment, V^T rows as B-fragments, pipelined ----
      const __bf16* pp = &Plds[wav][lq][half * 8];
      v16bf pa = frag16(pp, pp + 16);
      auto vfrag = [&](int d) {
        const __bf16* p = &VT[(16 * d + lq) * KV_TILE + half * 8];
        return frag16(p, p + 16);
      };
      v16bf vb = vfrag(0);
#pragma unroll
      for (int d = 0; d < 8; ++d) {
        v16bf nxt = vb;
        if (d < 7) nxt = vfrag(d + 1);             // issue next loads early
        oacc[d] = __builtin_amdgcn_wmma_f32_16x16x32_bf16(false, pa, false, vb,
                                                          (short)0, oacc[d], false, false);
        vb = nxt;
      }
    }
  }

  // ---- normalize and store (fp32 out) ----
  float* orow = out + gbase + (size_t)(qbase + r0) * DIMD;
#pragma unroll
  for (int r = 0; r < 8; ++r) {
    float inv = 1.0f / rs[r];
#pragma unroll
    for (int d = 0; d < 8; ++d)
      orow[(size_t)r * DIMD + 16 * d + lq] = oacc[d][r] * inv;
  }
}

extern "C" void kernel_launch(void* const* d_in, const int* in_sizes, int n_in,
                              void* d_out, int out_size, void* d_ws, size_t ws_size,
                              hipStream_t stream) {
  (void)n_in; (void)out_size; (void)d_ws; (void)ws_size;
  const float* q = (const float*)d_in[0];
  const float* k = (const float*)d_in[1];
  const float* v = (const float*)d_in[2];
  float* out = (float*)d_out;

  const int total   = in_sizes[0];                  // B*S*D
  const int nGroups = total / (GRP * DIMD);         // 8
  const int nBlocks = nGroups * (GRP / ROWS_PER_BLOCK);

  fa_causal_grouped_bf16<<<dim3(nBlocks), dim3(128), 0, stream>>>(q, k, v, out);
}
